// Point_contextual_hyper_prior_decoder_6047313953086
// MI455X (gfx1250) — matrix operand order repaired and checked by reading, not compiled, not run on hardware
//
#include <hip/hip_runtime.h>

typedef __attribute__((ext_vector_type(16))) __bf16 v16bf;
typedef __attribute__((ext_vector_type(8)))  float  v8f;

union FragU { v16bf v; uint4 q[2]; };

__device__ __forceinline__ unsigned short f2bf(float x){
  unsigned int u = __float_as_uint(x);
  unsigned int r = u + 0x7FFFu + ((u >> 16) & 1u);   // round-to-nearest-even
  return (unsigned short)(r >> 16);
}
__device__ __forceinline__ float bf2f(unsigned short h){
  return __uint_as_float(((unsigned int)h) << 16);
}

__device__ __forceinline__ v8f wmma_bf16(v16bf a, v16bf b, v8f c){
  // D(f32,16x16) = A(bf16,16x32) * B(bf16,32x16) + C
  return __builtin_amdgcn_wmma_f32_16x16x32_bf16(false, a, false, b, (short)0, c, false, false);
}

// A fragment (16x32 bf16, row = gathered voxel). ISA layout:
// lanes 0-15: VGPR0-3 = K 0..7, VGPR4-7 = K 16..23 ; lanes 16-31: K 8..15 / 24..31.
template<int CIN>
__device__ __forceinline__ v16bf load_a_frag(const unsigned short* __restrict__ base,
                                             size_t rowOff, int kbase, int lane){
  const int half = (lane >> 4) & 1;
  FragU f;
  if constexpr (CIN == 16) {
    f.q[0] = *(const uint4*)(base + rowOff + half * 8);        // K 0..15 valid
    f.q[1] = make_uint4(0u, 0u, 0u, 0u);                       // K 16..31 zero
  } else {
    const unsigned short* p = base + rowOff + kbase + half * 8;
    f.q[0] = *(const uint4*)(p);        // K kbase+half*8 .. +7
    f.q[1] = *(const uint4*)(p + 16);   // K kbase+16+half*8 .. +7
  }
  return f.v;
}

// B fragment (32x16 bf16). Weights stored transposed: [cout][cin] row-major.
// lanes 0-15 hold rows K=0..7,16..23 of column n=lane; lanes 16-31 hold K=8..15,24..31.
template<int CIN>
__device__ __forceinline__ v16bf load_b_frag(const unsigned short* __restrict__ wt,
                                             int kbase, int lane){
  const int n = lane & 15;
  const int khalf = (lane >> 4) & 1;
  FragU f;
  if constexpr (CIN == 16) {
    {
      const uint4* p = (const uint4*)(wt + n * 16 + khalf * 8);
      f.q[0] = p[0];
      f.q[1] = make_uint4(0u,0u,0u,0u);
    }
  } else {
    const uint4* p = (const uint4*)(wt + (size_t)n * CIN + kbase + khalf * 8);
    f.q[0] = p[0]; f.q[1] = p[2];
  }
  return f.v;
}

// Generic tile kernel with M-blocking: each wave owns MR x 16 rows x COUT cols.
// Sparse conv (GATHER, 27 taps) or dense GEMM (1 tap).
// fin: [rows(+pad), CIN] bf16; wt: [taps][COUT][CIN] bf16; resid: [M,64] bf16.
template<int CIN, int COUT, bool GATHER, bool RELU, bool RESID, int MR>
__global__ void __launch_bounds__(256) gemm_tile_kernel(
    const unsigned short* __restrict__ fin,
    const int* __restrict__ nbr,
    const unsigned short* __restrict__ wt,
    const float* __restrict__ bias,
    const unsigned short* __restrict__ resid,
    unsigned short* __restrict__ fout,
    int ldout, int outcol0)
{
  constexpr int NT   = COUT / 16;
  constexpr int KT   = (CIN == 16) ? 1 : (CIN / 32);
  constexpr int TAPS = GATHER ? 27 : 1;
  const int lane = threadIdx.x & 31;
  const int wave = threadIdx.x >> 5;
  const int m0   = (blockIdx.x * 8 + wave) * (16 * MR);

  v8f acc[MR][NT] = {};

  for (int k = 0; k < TAPS; ++k) {
    size_t rowOff[MR];
    #pragma unroll
    for (int j = 0; j < MR; ++j) {
      int arow = m0 + j * 16 + (lane & 15);
      int g = GATHER ? nbr[(size_t)arow * 27 + k] : arow;   // sentinel -> zeroed pad row
      rowOff[j] = (size_t)g * CIN;
    }
    #pragma unroll
    for (int kt = 0; kt < KT; ++kt) {
      v16bf A[MR];
      #pragma unroll
      for (int j = 0; j < MR; ++j)
        A[j] = load_a_frag<CIN>(fin, rowOff[j], kt * 32, lane);
      #pragma unroll
      for (int t = 0; t < NT; ++t) {
        v16bf B = load_b_frag<CIN>(wt + (size_t)(k * COUT + t * 16) * CIN, kt * 32, lane);
        #pragma unroll
        for (int j = 0; j < MR; ++j)
          acc[j][t] = wmma_bf16(A[j], B, acc[j][t]);   // B reused across MR row tiles
      }
    }
  }

  // D layout: lane -> col n (0..15), VGPR r -> row r (+8 for lanes 16..31)
  const int n = lane & 15;
  const int half = lane >> 4;
  #pragma unroll
  for (int j = 0; j < MR; ++j) {
    #pragma unroll
    for (int t = 0; t < NT; ++t) {
      #pragma unroll
      for (int r = 0; r < 8; ++r) {
        int row = m0 + j * 16 + r + 8 * half;
        int col = outcol0 + t * 16 + n;
        float v = acc[j][t][r] + bias[t * 16 + n];
        if constexpr (RELU)  v = v > 0.f ? v : 0.f;
        if constexpr (RESID) v += bf2f(resid[(size_t)row * 64 + col]);
        fout[(size_t)row * ldout + col] = f2bf(v);
      }
    }
  }
}

// Generative upsample: 8 dense GEMMs f1[N,64] @ wup[d][64,64], scatter rows 8n+d.
// MR=2 row blocking here too (B reused across both row tiles).
__global__ void __launch_bounds__(256) upsample_kernel(
    const unsigned short* __restrict__ f1,
    const unsigned short* __restrict__ wupT,
    const float* __restrict__ bup,
    unsigned short* __restrict__ fout)
{
  const int lane = threadIdx.x & 31;
  const int wave = threadIdx.x >> 5;
  const int n0 = (blockIdx.x * 8 + wave) * 32;
  const int d  = blockIdx.y;

  v8f acc[2][4] = {};
  #pragma unroll
  for (int kt = 0; kt < 2; ++kt) {
    v16bf A[2];
    #pragma unroll
    for (int j = 0; j < 2; ++j)
      A[j] = load_a_frag<64>(f1, (size_t)(n0 + j * 16 + (lane & 15)) * 64, kt * 32, lane);
    #pragma unroll
    for (int t = 0; t < 4; ++t) {
      v16bf B = load_b_frag<64>(wupT + (size_t)(d * 64 + t * 16) * 64, kt * 32, lane);
      #pragma unroll
      for (int j = 0; j < 2; ++j)
        acc[j][t] = wmma_bf16(A[j], B, acc[j][t]);
    }
  }
  const int n = lane & 15, half = lane >> 4;
  #pragma unroll
  for (int j = 0; j < 2; ++j) {
    #pragma unroll
    for (int t = 0; t < 4; ++t) {
      #pragma unroll
      for (int r = 0; r < 8; ++r) {
        int prow = n0 + j * 16 + r + 8 * half;
        int col  = t * 16 + n;
        float v = acc[j][t][r] + bup[col];
        v = v > 0.f ? v : 0.f;
        fout[((size_t)prow * 8 + d) * 64 + col] = f2bf(v);
      }
    }
  }
}

// conv1: tiny (1.8 GFLOP) — scalar with LDS-staged gathered inputs.
__global__ void __launch_bounds__(64) conv1_kernel(
    const float* __restrict__ z, const int* __restrict__ nbr_c,
    const float* __restrict__ w1, const float* __restrict__ b1,
    unsigned short* __restrict__ f1, int N)
{
  __shared__ float zs[216];
  const int n = blockIdx.x;
  const int tid = threadIdx.x;
  for (int t = tid; t < 216; t += 64) {
    int k = t >> 3, i = t & 7;
    int j = nbr_c[(size_t)n * 27 + k];
    zs[t] = (j < N) ? z[(size_t)j * 8 + i] : 0.f;
  }
  __syncthreads();
  float acc = b1[tid];
  for (int k = 0; k < 27; ++k) {
    #pragma unroll
    for (int i = 0; i < 8; ++i)
      acc += zs[k * 8 + i] * w1[(size_t)(k * 8 + i) * 64 + tid];
  }
  acc = acc > 0.f ? acc : 0.f;
  f1[(size_t)n * 64 + tid] = f2bf(acc);
}

// f32 [K,Ci,Co] -> bf16 [K,Co,Ci] (transpose for B fragments)
__global__ void prep_w_kernel(const float* __restrict__ src, unsigned short* __restrict__ dst,
                              int K, int Ci, int Co){
  size_t tot = (size_t)K * Ci * Co;
  size_t e = (size_t)blockIdx.x * 256 + threadIdx.x;
  if (e >= tot) return;
  int ci = (int)(e % Ci);
  size_t r = e / Ci;
  int co = (int)(r % Co);
  int k  = (int)(r / Co);
  dst[e] = f2bf(src[((size_t)k * Ci + ci) * Co + co]);
}

__global__ void zero_pads_kernel(unsigned short* fA, unsigned short* fB,
                                 unsigned short* t0, unsigned short* t1, int M){
  int t = threadIdx.x;
  if (t < 64) { fA[(size_t)M * 64 + t] = 0; fB[(size_t)M * 64 + t] = 0; }
  if (t < 16) { t0[(size_t)M * 16 + t] = 0; t1[(size_t)M * 16 + t] = 0; }
}

__global__ void to_f32_kernel(const unsigned short* __restrict__ f, float* __restrict__ out, size_t n){
  size_t i = (size_t)blockIdx.x * 256 + threadIdx.x;
  if (i < n) out[i] = bf2f(f[i]);
}

extern "C" void kernel_launch(void* const* d_in, const int* in_sizes, int n_in,
                              void* d_out, int out_size, void* d_ws, size_t ws_size,
                              hipStream_t stream) {
  const float* z     = (const float*)d_in[0];
  const int*   nbr_c = (const int*)d_in[1];
  const int*   nbr_f = (const int*)d_in[2];
  const float* w1    = (const float*)d_in[3];
  const float* b1    = (const float*)d_in[4];
  const float* wup   = (const float*)d_in[5];
  const float* bup   = (const float*)d_in[6];
  const float* w2    = (const float*)d_in[7];
  const float* b2    = (const float*)d_in[8];
  const float* bw0a  = (const float*)d_in[9];
  const float* bb0a  = (const float*)d_in[10];
  const float* bw0b  = (const float*)d_in[11];
  const float* bb0b  = (const float*)d_in[12];
  const float* bw1a  = (const float*)d_in[13];
  const float* bb1a  = (const float*)d_in[14];
  const float* bw1b  = (const float*)d_in[15];
  const float* bb1b  = (const float*)d_in[16];
  const float* bw1c  = (const float*)d_in[17];
  const float* bb1c  = (const float*)d_in[18];
  (void)n_in; (void)out_size; (void)ws_size;

  const int N = in_sizes[0] / 8;
  const int M = in_sizes[2] / 27;     // 8*N

  char* ws = (char*)d_ws;
  size_t off = 0;
  auto alloc = [&](size_t bytes) -> void* {
    void* p = ws + off;
    off = (off + bytes + 255) & ~(size_t)255;
    return p;
  };
  unsigned short* wupT  = (unsigned short*)alloc((size_t)8*64*64*2);
  unsigned short* w2T   = (unsigned short*)alloc((size_t)27*64*64*2);
  unsigned short* bw0aT = (unsigned short*)alloc((size_t)3*27*16*64*2);
  unsigned short* bw0bT = (unsigned short*)alloc((size_t)3*27*32*16*2);
  unsigned short* bw1aT = (unsigned short*)alloc((size_t)3*16*64*2);
  unsigned short* bw1bT = (unsigned short*)alloc((size_t)3*27*16*16*2);
  unsigned short* bw1cT = (unsigned short*)alloc((size_t)3*32*16*2);
  unsigned short* f1    = (unsigned short*)alloc((size_t)N*64*2);
  unsigned short* fA    = (unsigned short*)alloc(((size_t)M+1)*64*2);
  unsigned short* fB    = (unsigned short*)alloc(((size_t)M+1)*64*2);
  unsigned short* T0    = (unsigned short*)alloc(((size_t)M+1)*16*2);
  unsigned short* T1    = (unsigned short*)alloc(((size_t)M+1)*16*2);

  auto prep = [&](const float* src, unsigned short* dst, int K, int Ci, int Co){
    size_t tot = (size_t)K*Ci*Co;
    prep_w_kernel<<<dim3((unsigned)((tot+255)/256)), dim3(256), 0, stream>>>(src, dst, K, Ci, Co);
  };
  prep(wup,  wupT,  8, 64, 64);
  prep(w2,   w2T,  27, 64, 64);
  prep(bw0a, bw0aT, 81, 64, 16);   // 3*27 taps flattened
  prep(bw0b, bw0bT, 81, 16, 32);
  prep(bw1a, bw1aT,  3, 64, 16);
  prep(bw1b, bw1bT, 81, 16, 16);
  prep(bw1c, bw1cT,  3, 16, 32);

  zero_pads_kernel<<<dim3(1), dim3(64), 0, stream>>>(fA, fB, T0, T1, M);

  conv1_kernel<<<dim3((unsigned)N), dim3(64), 0, stream>>>(z, nbr_c, w1, b1, f1, N);

  // 8 waves/block, 32 parent rows per wave -> N/256 blocks in x, d in y
  upsample_kernel<<<dim3((unsigned)(N/256), 8), dim3(256), 0, stream>>>(f1, wupT, bup, fA);

  constexpr int MR = 2;
  dim3 gtile((unsigned)(M / (16 * MR * 8)));
  // conv2: fA -> fB (no relu)
  gemm_tile_kernel<64,64,true,false,false,MR><<<gtile,256,0,stream>>>(fA, nbr_f, w2T, b2, nullptr, fB, 64, 0);

  unsigned short* fcur = fB;
  unsigned short* fnext = fA;
  for (int i = 0; i < 3; ++i) {
    // t0 = relu(sconv(f, bw0a[i]))               -> T0
    gemm_tile_kernel<64,16,true,true,false,MR><<<gtile,256,0,stream>>>(
        fcur, nbr_f, bw0aT + (size_t)i*27*16*64, bb0a + i*16, nullptr, T0, 16, 0);
    // t1 = relu(f @ bw1a[i])                     -> T1
    gemm_tile_kernel<64,16,false,true,false,MR><<<gtile,256,0,stream>>>(
        fcur, nbr_f, bw1aT + (size_t)i*16*64, bb1a + i*16, nullptr, T1, 16, 0);
    // fnext[:,0:32] = sconv(T0, bw0b[i]) + f[:,0:32]
    gemm_tile_kernel<16,32,true,false,true,MR><<<gtile,256,0,stream>>>(
        T0, nbr_f, bw0bT + (size_t)i*27*32*16, bb0b + i*32, fcur, fnext, 64, 0);
    // t2 = relu(sconv(T1, bw1b[i]))              -> T0 (reuse)
    gemm_tile_kernel<16,16,true,true,false,MR><<<gtile,256,0,stream>>>(
        T1, nbr_f, bw1bT + (size_t)i*27*16*16, bb1b + i*16, nullptr, T0, 16, 0);
    // fnext[:,32:64] = T0 @ bw1c[i] + f[:,32:64]
    gemm_tile_kernel<16,32,false,false,true,MR><<<gtile,256,0,stream>>>(
        T0, nbr_f, bw1cT + (size_t)i*32*16, bb1c + i*32, fcur, fnext, 64, 32);
    unsigned short* tmp = fcur; fcur = fnext; fnext = tmp;
  }

  size_t tot = (size_t)M * 64;
  to_f32_kernel<<<dim3((unsigned)((tot+255)/256)), 256, 0, stream>>>(fcur, (float*)d_out, tot);
}
